// GCN_pre_define_20667382628531
// MI455X (gfx1250) — matrix operand order repaired and checked by reading, not compiled, not run on hardware
//
#include <hip/hip_runtime.h>
#include <hip/hip_bf16.h>

// ---------- types ----------
typedef __attribute__((ext_vector_type(16))) _Float16 v16h;
typedef __attribute__((ext_vector_type(8)))  float    v8f;
typedef __attribute__((ext_vector_type(4)))  unsigned int v4u;

#define N_NODES 1601
#define EMB     300
#define EMB_P   320      // padded to 10 K-tiles of 32
#define BATCH   2048
#define L_SET   64

// LDS layout (bytes), all 16B aligned
#define OFF_LAB   0        // 64 * int            = 256
#define OFF_DINV  256      // 64 * float          = 256
#define OFF_ADJ   512      // 64 x 72 f16         = 9216
#define OFF_G     9728     // 320 x 72 f16        = 46080  (h, E-major: G[e][i])
#define OFF_P     55808    // 64 x 328 f16        = 41984  (P, node-major: P[i][e])
#define OFF_NB    55808    // 64 x 68 f32 = 17408 (aliases P; setup only)
#define SMEM_BYTES 97792

#define PITCH_G   72
#define PITCH_P   328
#define PITCH_ADJ 72
#define PITCH_NB  68

__device__ __forceinline__ v16h load_frag_pair(const _Float16* p0, const _Float16* p1) {
  union { v4u q[2]; v16h h; } u;
  u.q[0] = *(const v4u*)p0;
  u.q[1] = *(const v4u*)p1;
  return u.h;
}

__device__ __forceinline__ v8f wmma16(v16h a, v16h b, v8f c) {
  // D = A*B + C, f16 inputs, f32 accumulate
  return __builtin_amdgcn_wmma_f32_16x16x32_f16(false, a, false, b, (short)0, c,
                                                false, false);
}

// store D-fragment transposed: lane holds column n, rows m = 8*hi + 0..7
__device__ __forceinline__ void store_ct(const v8f& acc, _Float16* dst) {
  union { v4u q; _Float16 h[8]; } u;
#pragma unroll
  for (int r = 0; r < 8; ++r) u.h[r] = (_Float16)acc[r];
  *(v4u*)dst = u.q;
}

// -------------------------------------------------------------------------
// Prep: pack linear_w (3,300,300) into B-fragment order, f16, padded.
// Tile index = l*400 + nt*20 + kt (kt slots 10..19 are zeros: harmless
// prefetch targets). dword d = tile*256 + lane*8 + j2 ; dword = f16 pair.
// -------------------------------------------------------------------------
#define WF_DWORDS (3 * 20 * 20 * 32 * 8)   // 307200 dwords = 1.2 MB

__global__ __launch_bounds__(256) void gcn_prep_w(const float* __restrict__ W,
                                                  unsigned int* __restrict__ wf) {
  int d = blockIdx.x * 256 + threadIdx.x;
  if (d >= WF_DWORDS) return;
  int j2   = d & 7;
  int lane = (d >> 3) & 31;
  int tile = d >> 8;
  int kt = tile % 20;
  int nt = (tile / 20) % 20;
  int l  = tile / 400;
  int e_out = nt * 16 + (lane & 15);
  int e_in  = kt * 32 + ((j2 >> 2) << 4) + ((lane >> 4) << 3) + ((j2 & 3) << 1);
  float f0 = 0.f, f1 = 0.f;
  if (e_out < EMB) {
    const float* wr = W + l * (EMB * EMB) + e_out * EMB;
    if (e_in     < EMB) f0 = wr[e_in];
    if (e_in + 1 < EMB) f1 = wr[e_in + 1];
  }
  union { _Float16 h[2]; unsigned int u; } cv;
  cv.h[0] = (_Float16)f0;
  cv.h[1] = (_Float16)f1;
  wf[d] = cv.u;
}

// -------------------------------------------------------------------------
// Main kernel: one workgroup (256 threads = 8 waves) per batch element.
// -------------------------------------------------------------------------
__global__ __launch_bounds__(256) void gcn_layers(
    const int* __restrict__ labels, const float* __restrict__ in_adj,
    const float* __restrict__ emb, const unsigned int* __restrict__ wf,
    float* __restrict__ out) {
  extern __shared__ char smem[];
  int*       lab  = (int*)(smem + OFF_LAB);
  float*     dinv = (float*)(smem + OFF_DINV);
  _Float16*  adjL = (_Float16*)(smem + OFF_ADJ);
  _Float16*  G    = (_Float16*)(smem + OFF_G);
  _Float16*  P    = (_Float16*)(smem + OFF_P);
  float*     nb   = (float*)(smem + OFF_NB);

  const int t    = threadIdx.x;
  const int b    = blockIdx.x;
  const int wave = t >> 5;
  const int lane = t & 31;
  const int mrow = lane & 15;
  const int hi   = lane >> 4;

  // ---- setup: labels ----
  if (t < 64) lab[t] = labels[b * L_SET + t];
  __syncthreads();

  // ---- neighbor gather (+I) ----
  for (int p = t; p < 64 * 64; p += 256) {
    int i = p >> 6, j = p & 63;
    float v = in_adj[lab[i] * N_NODES + lab[j]];
    if (i == j) v += 1.0f;
    nb[i * PITCH_NB + j] = v;
  }
  __syncthreads();

  // ---- column sums -> d^-1/2 ----
  if (t < 64) {
    float s = 0.f;
    for (int i = 0; i < 64; ++i) s += nb[i * PITCH_NB + t];
    dinv[t] = rsqrtf(s);
  }
  __syncthreads();

  // ---- adj (f16, row-major) ----
  for (int p = t; p < 64 * 64; p += 256) {
    int i = p >> 6, j = p & 63;
    adjL[i * PITCH_ADJ + j] = (_Float16)(nb[i * PITCH_NB + j] * dinv[i] * dinv[j]);
  }

  // ---- h0: embeddings gather into G (E-major: G[e][i] = emb[lab_i][e]) ----
  // consecutive lanes -> consecutive e of one embedding row (coalesced)
  for (int p = t; p < 64 * EMB_P; p += 256) {
    int i = p / EMB_P;
    int e = p - i * EMB_P;
    float v = (e < EMB) ? emb[lab[i] * EMB + e] : 0.f;
    G[e * PITCH_G + i] = (_Float16)v;
  }
  __syncthreads();

  const v8f vzero = {0.f, 0.f, 0.f, 0.f, 0.f, 0.f, 0.f, 0.f};

  for (int l = 0; l < 3; ++l) {
    // adj B-fragments are invariant within the layer: hoist into registers.
    // badj[2*nt + kt]:  B[k=j][n=i] = adj[i][j], lane = row i of adj.
    v16h badj[8];
#pragma unroll
    for (int nt = 0; nt < 4; ++nt) {
      const _Float16* bp = adjL + (nt * 16 + mrow) * PITCH_ADJ + 8 * hi;
      badj[2 * nt + 0] = load_frag_pair(bp, bp + 16);
      badj[2 * nt + 1] = load_frag_pair(bp + 32, bp + 48);
    }

    // ======== Phase A:  P^T = h^T * adj^T (M=e tiles, N=node tiles, K=64)
    for (int mtE = wave; mtE < 20; mtE += 8) {
      const _Float16* ga = G + (mtE * 16 + mrow) * PITCH_G + 8 * hi;
      v16h a0 = load_frag_pair(ga, ga + 16);        // K-tile 0
      v16h a1 = load_frag_pair(ga + 32, ga + 48);   // K-tile 1
      v8f acc[4];
#pragma unroll
      for (int nt = 0; nt < 4; ++nt) {
        acc[nt] = wmma16(a0, badj[2 * nt + 0], vzero);
        acc[nt] = wmma16(a1, badj[2 * nt + 1], acc[nt]);
      }
#pragma unroll
      for (int nt = 0; nt < 4; ++nt)  // D^T store: P node-major
        store_ct(acc[nt], P + (nt * 16 + mrow) * PITCH_P + mtE * 16 + 8 * hi);
    }
    __syncthreads();

    // ======== Phase B:  h_new = P * W^T (M=node, N=e_out tiles, K=320)
    for (int nt = wave; nt < 20; nt += 8) {
      v8f acc[4];
#pragma unroll
      for (int mt = 0; mt < 4; ++mt) acc[mt] = vzero;

      const unsigned int* wb = wf + ((l * 400 + nt * 20) << 8) + lane * 8;
      union BU { v4u q[2]; v16h h; };
      BU b0, b1;
      b0.q[0] = *(const v4u*)(wb);            // prefetch kt = 0
      b0.q[1] = *(const v4u*)(wb + 4);
#pragma unroll
      for (int kt = 0; kt < 10; kt += 2) {
        // prefetch kt+1 (always valid)
        b1.q[0] = *(const v4u*)(wb + (kt + 1) * 256);
        b1.q[1] = *(const v4u*)(wb + (kt + 1) * 256 + 4);
#pragma unroll
        for (int mt = 0; mt < 4; ++mt) {
          const _Float16* ap = P + (mt * 16 + mrow) * PITCH_P + kt * 32 + 8 * hi;
          acc[mt] = wmma16(load_frag_pair(ap, ap + 16), b0.h, acc[mt]);
        }
        // prefetch kt+2 (slots up to 19 exist; 10.. are zeros -> safe)
        b0.q[0] = *(const v4u*)(wb + (kt + 2) * 256);
        b0.q[1] = *(const v4u*)(wb + (kt + 2) * 256 + 4);
#pragma unroll
        for (int mt = 0; mt < 4; ++mt) {
          const _Float16* ap =
              P + (mt * 16 + mrow) * PITCH_P + (kt + 1) * 32 + 8 * hi;
          acc[mt] = wmma16(load_frag_pair(ap, ap + 16), b1.h, acc[mt]);
        }
      }
#pragma unroll
      for (int mt = 0; mt < 4; ++mt)  // D^T store: G E-major
        store_ct(acc[mt], G + (nt * 16 + mrow) * PITCH_G + mt * 16 + 8 * hi);
    }
    __syncthreads();
  }

  // ---- epilogue: out[b][i][e] = G[e][i] (f32, streaming / non-temporal) ----
  for (int p = t; p < 64 * EMB; p += 256) {
    int i = p / EMB;
    int e = p - i * EMB;
    float v = (float)G[e * PITCH_G + i];
    __builtin_nontemporal_store(v, &out[(b * L_SET + i) * EMB + e]);
  }
}

// -------------------------------------------------------------------------
extern "C" void kernel_launch(void* const* d_in, const int* in_sizes, int n_in,
                              void* d_out, int out_size, void* d_ws, size_t ws_size,
                              hipStream_t stream) {
  const int*   labels = (const int*)d_in[0];
  const float* in_adj = (const float*)d_in[1];
  const float* emb    = (const float*)d_in[2];
  const float* lw     = (const float*)d_in[3];
  unsigned int* wf    = (unsigned int*)d_ws;
  float* out          = (float*)d_out;

  if (ws_size < (size_t)WF_DWORDS * 4) return;  // workspace too small: bail safely

  gcn_prep_w<<<(WF_DWORDS + 255) / 256, 256, 0, stream>>>(lw, wf);
  gcn_layers<<<BATCH, 256, SMEM_BYTES, stream>>>(labels, in_adj, emb, wf, out);
}
